// MoEAdaptorLayer_18580028523117
// MI455X (gfx1250) — matrix operand order repaired and checked by reading, not compiled, not run on hardware
//
#include <hip/hip_runtime.h>
#include <hip/hip_bf16.h>
#include <stdint.h>
#include <stddef.h>

// ---------------- types ----------------
typedef __bf16 bf16;
typedef __attribute__((ext_vector_type(16))) __bf16 bf16x16;
typedef __attribute__((ext_vector_type(8)))  float  f32x8;

union Frag {
  bf16x16 v;
  uint4   q[2];
};

// ---------------- problem dims ----------------
constexpr int Bb  = 64;
constexpr int Tt  = 300;
constexpr int DIN = 768;
constexpr int Dd  = 300;
constexpr int Ee  = 8;
constexpr int Hh  = 4;
constexpr int Mrows = Bb * Tt;       // 19200 (divisible by 128)
constexpr int DPAD  = 320;           // K pad: 300 -> 320 (mult of 32)
constexpr int HPAD  = 608;           // K pad: 600 -> 608 (mult of 32)

// ---------------- GEMM tile config ----------------
#define BM 128
#define BN 64
#define TPB 256

// MODE: 0 = fp32 C; 1 = bf16 Cb (zero-padded to ldcb); 2 = both;
//       3 = MoE accumulate: S1 += g*val, S2 += val (init stores).
// B-matrix rows must be zero-padded up to gridDim.y*BN (async loads can't mask).
template <bool RELU, int MODE>
__global__ __launch_bounds__(TPB) void gemm_bf16_wmma(
    const bf16* __restrict__ A,  const bf16* __restrict__ Bw,
    const float* __restrict__ bias,
    float* __restrict__ C, int ldc,
    bf16* __restrict__ Cb, int ldcb,
    const float* __restrict__ gate, int initAcc,
    float* __restrict__ S2,
    int M, int N, int K)
{
  __shared__ __align__(16) bf16 As[2][BM * 32];   // 2 x 8 KB
  __shared__ __align__(16) bf16 Bs[2][BN * 32];   // 2 x 4 KB

  const int tid  = threadIdx.x;
  const int lane = tid & 31;
  const int wave = tid >> 5;
  const int wm   = wave & 3;      // wave row group (x32 rows)
  const int wn   = wave >> 2;     // wave col group (x32 cols)
  const int hf   = lane >> 4;     // lane half
  const int lr   = lane & 15;

  const int tileM = blockIdx.x * BM;
  const int tileN = blockIdx.y * BN;

  const f32x8 fzero = {0.f, 0.f, 0.f, 0.f, 0.f, 0.f, 0.f, 0.f};
  f32x8 acc[2][2];
  acc[0][0] = fzero; acc[0][1] = fzero; acc[1][0] = fzero; acc[1][1] = fzero;

  // Per-thread 16B chunk coordinates for tile staging.
  const int ra = tid >> 2;            // A rows: ra and ra+64
  const int ca = (tid & 3) * 8;       // A k-offset (elements)
  const int rb = tid >> 2;            // B row (0..63)
  const int cb = (tid & 3) * 8;

  // Async-copy one 32-deep K slab into LDS buffer `buf` (ASYNCcnt += 3/wave).
  auto issue = [&](int k0, int buf) {
    const bf16* ga0 = A  + (size_t)(tileM + ra)      * K + (k0 + ca);
    const bf16* ga1 = A  + (size_t)(tileM + ra + 64) * K + (k0 + ca);
    const bf16* gb  = Bw + (size_t)(tileN + rb)      * K + (k0 + cb);
    uint32_t la0 = (uint32_t)(uintptr_t)(&As[buf][ra * 32 + ca]);
    uint32_t la1 = (uint32_t)(uintptr_t)(&As[buf][(ra + 64) * 32 + ca]);
    uint32_t lb  = (uint32_t)(uintptr_t)(&Bs[buf][rb * 32 + cb]);
    asm volatile("global_load_async_to_lds_b128 %0, %1, off" :: "v"(la0), "v"(ga0) : "memory");
    asm volatile("global_load_async_to_lds_b128 %0, %1, off" :: "v"(la1), "v"(ga1) : "memory");
    asm volatile("global_load_async_to_lds_b128 %0, %1, off" :: "v"(lb),  "v"(gb)  : "memory");
  };

  const int nk = K >> 5;
  issue(0, 0);

  for (int t = 0; t < nk; ++t) {
    const int buf = t & 1;
    if (t + 1 < nk) {
      issue((t + 1) * 32, buf ^ 1);                       // prefetch next slab
      asm volatile("s_wait_asynccnt 0x3" ::: "memory");   // oldest 3 (this slab) done
    } else {
      asm volatile("s_wait_asynccnt 0x0" ::: "memory");
    }
    __syncthreads();   // all waves' async writes for `buf` visible

    const bf16* Asb = &As[buf][0];
    const bf16* Bsb = &Bs[buf][0];
    Frag af[2], bfg[2];
#pragma unroll
    for (int i = 0; i < 2; ++i) {
      int m = wm * 32 + i * 16 + lr;
      af[i].q[0] = *(const uint4*)(Asb + m * 32 + hf * 8);        // K = hf*8 .. +7
      af[i].q[1] = *(const uint4*)(Asb + m * 32 + 16 + hf * 8);   // K = 16+hf*8 .. +7
      int n = wn * 32 + i * 16 + lr;
      bfg[i].q[0] = *(const uint4*)(Bsb + n * 32 + hf * 8);       // K = hf*8 .. +7
      bfg[i].q[1] = *(const uint4*)(Bsb + n * 32 + 16 + hf * 8);  // K = 16+hf*8 .. +7
    }
#pragma unroll
    for (int i = 0; i < 2; ++i)
#pragma unroll
      for (int j = 0; j < 2; ++j)
        acc[i][j] = __builtin_amdgcn_wmma_f32_16x16x32_bf16(
            false, af[i].v, false, bfg[j].v, (short)0, acc[i][j], false, false);
    __syncthreads();   // reads of `buf` done before anyone refills it
  }

  // Epilogue
#pragma unroll
  for (int i = 0; i < 2; ++i) {
#pragma unroll
    for (int j = 0; j < 2; ++j) {
      int n = tileN + wn * 32 + j * 16 + lr;
      float bv = (bias != nullptr && n < N) ? bias[n] : 0.f;
#pragma unroll
      for (int r = 0; r < 8; ++r) {
        int m = tileM + wm * 32 + i * 16 + hf * 8 + r;
        float val = acc[i][j][r] + bv;
        if (RELU) val = fmaxf(val, 0.f);
        if constexpr (MODE == 0 || MODE == 2) {
          if (n < N) C[(size_t)m * ldc + n] = val;
        }
        if constexpr (MODE == 1 || MODE == 2) {
          if (n < ldcb) Cb[(size_t)m * ldcb + n] = (n < N) ? (bf16)val : (bf16)0.f;
        }
        if constexpr (MODE == 3) {
          if (n < N) {
            float gr = gate[(size_t)m * 8];
            size_t o = (size_t)m * ldc + n;
            if (initAcc) { C[o] = gr * val; S2[o] = val; }
            else         { C[o] += gr * val; S2[o] += val; }
          }
        }
      }
    }
  }
}

// ---------------- conversions ----------------
__global__ void conv_x_kernel(const float* __restrict__ x, const float* __restrict__ bias,
                              bf16* __restrict__ out, int total, int K) {
  int i = blockIdx.x * 256 + threadIdx.x;
  if (i < total) out[i] = (bf16)(x[i] - bias[i % K]);
}

// fp32 [Nrows,K] -> bf16 [Npad,Kpad], zero-filled rows >= Nrows and cols >= K.
__global__ void conv_w_kernel(const float* __restrict__ in, bf16* __restrict__ out,
                              int Nrows, int Npad, int K, int Kpad) {
  int i = blockIdx.x * 256 + threadIdx.x;
  int total = Npad * Kpad;
  if (i >= total) return;
  int r = i / Kpad, k = i - r * Kpad;
  out[i] = (r < Nrows && k < K) ? (bf16)in[(size_t)r * K + k] : (bf16)0.f;
}

// ---------------- fused causal attention (one wave per (b,h,t)) ----------------
__global__ __launch_bounds__(256) void attn_kernel(const float* __restrict__ qkv,
                                                   bf16* __restrict__ attb) {
  int gtid = blockIdx.x * blockDim.x + threadIdx.x;
  int wid  = gtid >> 5;
  int lane = threadIdx.x & 31;
  if (wid >= Bb * Hh * Tt) return;
  int b = wid / (Hh * Tt);
  int rem = wid - b * Hh * Tt;
  int h = rem / Tt;
  int t = rem - h * Tt;

  const float* qrow = qkv + ((size_t)(b * Tt + t)) * 192 + h * 16;
  float q[16];
#pragma unroll
  for (int j = 0; j < 16; ++j) q[j] = qrow[j];

  float m = -INFINITY, l = 0.f;
  float acc[16];
#pragma unroll
  for (int j = 0; j < 16; ++j) acc[j] = 0.f;

  const float scale = 0.05773502691896258f;  // 300^-0.5
  for (int s = lane; s <= t; s += 32) {
    const float* krow = qkv + ((size_t)(b * Tt + s)) * 192 + 64 + h * 16;
    const float* vrow = krow + 64;
    float d = 0.f;
#pragma unroll
    for (int j = 0; j < 16; ++j) d += q[j] * krow[j];
    d *= scale;
    float nm = fmaxf(m, d);
    float corr = __expf(m - nm);   // first iter: exp(-inf)=0
    float p = __expf(d - nm);
    l = l * corr + p;
#pragma unroll
    for (int j = 0; j < 16; ++j) acc[j] = acc[j] * corr + p * vrow[j];
    m = nm;
  }
  // cross-lane merge of (m, l, acc) — wave32 xor tree
#pragma unroll
  for (int off = 16; off > 0; off >>= 1) {
    float m2 = __shfl_xor(m, off);
    float l2 = __shfl_xor(l, off);
    float nm = fmaxf(m, m2);
    float w1 = (l  > 0.f) ? __expf(m  - nm) : 0.f;
    float w2 = (l2 > 0.f) ? __expf(m2 - nm) : 0.f;
#pragma unroll
    for (int j = 0; j < 16; ++j) {
      float a2 = __shfl_xor(acc[j], off);
      acc[j] = acc[j] * w1 + a2 * w2;
    }
    l = l * w1 + l2 * w2;
    m = nm;
  }
  if (lane == 0) {
    float inv = 1.f / l;
    bf16* o = attb + ((size_t)(b * Tt + t)) * 64 + h * 16;
#pragma unroll
    for (int j = 0; j < 16; ++j) o[j] = (bf16)(acc[j] * inv);
  }
}

// ---------------- noisy-top-k gating (eval): top-4 softmax ----------------
__global__ __launch_bounds__(256) void gate_kernel(const float* __restrict__ xa,
                                                   const float* __restrict__ wgate,
                                                   float* __restrict__ gprob) {
  int gtid = blockIdx.x * blockDim.x + threadIdx.x;
  int wid  = gtid >> 5;
  int lane = threadIdx.x & 31;
  if (wid >= Mrows) return;
  float lg[Ee];
#pragma unroll
  for (int e = 0; e < Ee; ++e) lg[e] = 0.f;
  const float* row = xa + (size_t)wid * Dd;
  for (int d = lane; d < Dd; d += 32) {
    float xv = row[d];
    const float* wg = wgate + d * Ee;
#pragma unroll
    for (int e = 0; e < Ee; ++e) lg[e] += xv * wg[e];
  }
#pragma unroll
  for (int off = 16; off > 0; off >>= 1)
#pragma unroll
    for (int e = 0; e < Ee; ++e) lg[e] += __shfl_xor(lg[e], off);

  if (lane == 0) {
    bool sel[Ee];
    float tmp[Ee];
#pragma unroll
    for (int e = 0; e < Ee; ++e) { sel[e] = false; tmp[e] = lg[e]; }
    for (int it = 0; it < Ee / 2; ++it) {  // TOP_K = 4
      int bi = 0; float bv = -INFINITY;
      for (int e = 0; e < Ee; ++e)
        if (!sel[e] && tmp[e] > bv) { bv = tmp[e]; bi = e; }
      sel[bi] = true;
    }
    float mx = -INFINITY;
    for (int e = 0; e < Ee; ++e) if (sel[e]) mx = fmaxf(mx, lg[e]);
    float p[Ee]; float s = 0.f;
    for (int e = 0; e < Ee; ++e) { p[e] = sel[e] ? __expf(lg[e] - mx) : 0.f; s += p[e]; }
    float inv = 1.f / s;
    float* o = gprob + (size_t)wid * Ee;
    for (int e = 0; e < Ee; ++e) o[e] = p[e] * inv;
  }
}

// ---------------- finalize: out = S1 * S2 ----------------
__global__ void finalize_kernel(const float* __restrict__ S1, const float* __restrict__ S2,
                                float* __restrict__ out, int n) {
  int i = blockIdx.x * 256 + threadIdx.x;
  if (i < n) out[i] = S1[i] * S2[i];
}

// ---------------- host ----------------
extern "C" void kernel_launch(void* const* d_in, const int* in_sizes, int n_in,
                              void* d_out, int out_size, void* d_ws, size_t ws_size,
                              hipStream_t stream) {
  (void)in_sizes; (void)n_in; (void)out_size; (void)ws_size;
  const float* x     = (const float*)d_in[0];
  const float* whb   = (const float*)d_in[1];
  const float* whW   = (const float*)d_in[2];
  const float* Wq    = (const float*)d_in[3];
  const float* Wk    = (const float*)d_in[4];
  const float* Wv    = (const float*)d_in[5];
  const float* projW = (const float*)d_in[6];
  const float* projb = (const float*)d_in[7];
  const float* eW1   = (const float*)d_in[8];
  const float* eb1   = (const float*)d_in[9];
  const float* eW2   = (const float*)d_in[10];
  const float* eb2   = (const float*)d_in[11];
  const float* wgate = (const float*)d_in[12];
  float* out = (float*)d_out;

  char* ws = (char*)d_ws;
  size_t off = 0;
  auto alloc = [&](size_t bytes) -> char* {
    char* p = ws + off;
    off = (off + bytes + 255) & ~(size_t)255;
    return p;
  };

  // B-side buffers padded to BN-tile multiples (rows) and 32-multiples (K).
  bf16*  xbf    = (bf16*)alloc((size_t)Mrows * DIN * 2);       // reused as hbf later
  bf16*  whWbf  = (bf16*)alloc((size_t)320 * DIN * 2);         // 300 -> 320 rows
  bf16*  xwbf   = (bf16*)alloc((size_t)Mrows * DPAD * 2);
  bf16*  wqkvbf = (bf16*)alloc((size_t)192 * DPAD * 2);        // 192 rows exact
  float* qkv    = (float*)alloc((size_t)Mrows * 192 * 4);
  bf16*  attbf  = (bf16*)alloc((size_t)Mrows * 64 * 2);
  bf16*  pWbf   = (bf16*)alloc((size_t)320 * 64 * 2);          // 300 -> 320 rows
  float* xa     = (float*)alloc((size_t)Mrows * Dd * 4);
  bf16*  xabf   = (bf16*)alloc((size_t)Mrows * DPAD * 2);
  float* gprob  = (float*)alloc((size_t)Mrows * Ee * 4);
  bf16*  W1bf   = (bf16*)alloc((size_t)Ee * 640 * DPAD * 2);   // 600 -> 640 rows/expert
  bf16*  W2bf   = (bf16*)alloc((size_t)Ee * 320 * HPAD * 2);   // 300 -> 320 rows/expert
  float* S1     = (float*)alloc((size_t)Mrows * Dd * 4);
  float* S2     = (float*)alloc((size_t)Mrows * Dd * 4);
  bf16*  hbf    = xbf;  // alias: 19200*608*2 (23.3MB) < 19200*768*2 (29.5MB)

  const int MGX = Mrows / BM;  // 150

  // 1) x - bias -> bf16
  {
    int total = Mrows * DIN;
    conv_x_kernel<<<(total + 255) / 256, 256, 0, stream>>>(x, whb, xbf, total, DIN);
  }
  // 2) weight conversions (row + K zero padding)
  conv_w_kernel<<<(320 * DIN + 255) / 256, 256, 0, stream>>>(whW, whWbf, Dd, 320, DIN, DIN);
  conv_w_kernel<<<(64 * DPAD + 255) / 256, 256, 0, stream>>>(Wq, wqkvbf,              64, 64, Dd, DPAD);
  conv_w_kernel<<<(64 * DPAD + 255) / 256, 256, 0, stream>>>(Wk, wqkvbf + 64 * DPAD,  64, 64, Dd, DPAD);
  conv_w_kernel<<<(64 * DPAD + 255) / 256, 256, 0, stream>>>(Wv, wqkvbf + 128 * DPAD, 64, 64, Dd, DPAD);
  conv_w_kernel<<<(320 * 64 + 255) / 256, 256, 0, stream>>>(projW, pWbf, Dd, 320, 64, 64);
  for (int e = 0; e < Ee; ++e) {
    conv_w_kernel<<<(640 * DPAD + 255) / 256, 256, 0, stream>>>(
        eW1 + (size_t)e * 600 * Dd, W1bf + (size_t)e * 640 * DPAD, 600, 640, Dd, DPAD);
    conv_w_kernel<<<(320 * HPAD + 255) / 256, 256, 0, stream>>>(
        eW2 + (size_t)e * Dd * 600, W2bf + (size_t)e * 320 * HPAD, Dd, 320, 600, HPAD);
  }

  // 3) whiten: xw = (x-b) @ whW^T  -> bf16 [M, 320]
  gemm_bf16_wmma<false, 1><<<dim3(MGX, 5), TPB, 0, stream>>>(
      xbf, whWbf, nullptr, nullptr, 0, xwbf, DPAD, nullptr, 0, nullptr, Mrows, Dd, DIN);

  // 4) qkv = xw @ [Wq;Wk;Wv]^T -> fp32 [M, 192]
  gemm_bf16_wmma<false, 0><<<dim3(MGX, 3), TPB, 0, stream>>>(
      xwbf, wqkvbf, nullptr, qkv, 192, nullptr, 0, nullptr, 0, nullptr, Mrows, 192, DPAD);

  // 5) fused causal attention -> att bf16 [M, 64]
  attn_kernel<<<(Bb * Hh * Tt) / 8, 256, 0, stream>>>(qkv, attbf);

  // 6) xa = att @ projW^T + b -> fp32 [M,300] + bf16 [M,320]
  gemm_bf16_wmma<false, 2><<<dim3(MGX, 5), TPB, 0, stream>>>(
      attbf, pWbf, projb, xa, Dd, xabf, DPAD, nullptr, 0, nullptr, Mrows, Dd, 64);

  // 7) top-4 gating probs
  gate_kernel<<<Mrows / 8, 256, 0, stream>>>(xa, wgate, gprob);

  // 8) experts: h = relu(xa@W1^T+b1) -> bf16 [M,608]; then gated accumulate
  for (int e = 0; e < Ee; ++e) {
    gemm_bf16_wmma<true, 1><<<dim3(MGX, 10), TPB, 0, stream>>>(
        xabf, W1bf + (size_t)e * 640 * DPAD, eb1 + (size_t)e * 600,
        nullptr, 0, hbf, HPAD, nullptr, 0, nullptr, Mrows, 600, DPAD);
    gemm_bf16_wmma<false, 3><<<dim3(MGX, 5), TPB, 0, stream>>>(
        hbf, W2bf + (size_t)e * 320 * HPAD, eb2 + (size_t)e * Dd,
        S1, Dd, nullptr, 0, gprob + e, (e == 0) ? 1 : 0, S2, Mrows, Dd, HPAD);
  }

  // 9) out = S1 * S2
  {
    int n = Mrows * Dd;
    finalize_kernel<<<(n + 255) / 256, 256, 0, stream>>>(S1, S2, out, n);
  }
}
